// BatteryMoEInterCycleMoELayer_25357486916138
// MI455X (gfx1250) — matrix operand order repaired and checked by reading, not compiled, not run on hardware
//
#include <hip/hip_runtime.h>
#include <hip/hip_bf16.h>
#include <math.h>

// ---------------------------------------------------------------------------
// BatteryMoE inter-cycle MoE layer for MI455X (gfx1250, wave32, WMMA).
// B=64, L=512, D=512, F=2048, E=8.  out = sum_e gates[b,e]*expert_e(x), plus
// guide_loss scalar appended at d_out[B*L*D].
//
// bf16 hi/lo split (3x v_wmma_f32_16x16x32_bf16 per 32-deep k-step) gives
// ~f32 accuracy at bf16 matrix-pipe rates. Weights pre-split + pre-swizzled
// into the WMMA B-operand lane layout in d_ws -> hot loop is pure
// global_load_b128 / ds_load_b128 + WMMA.  BM=32 rows/block: each weight
// tile load feeds 2 m-tiles (6 WMMAs), halving L2 weight traffic.
// Gate is folded into the GELU epilogue ((g*h)@W2), removing the per-expert
// y accumulator entirely.
// ---------------------------------------------------------------------------

typedef __attribute__((ext_vector_type(16))) __bf16 v16bf;
typedef __attribute__((ext_vector_type(8)))  float  v8f;

#define Bb 64
#define Ll 512
#define Dd 512
#define Ff 2048
#define Ee 8

#define NT1 (Ff/16)   // 128 n-tiles for GEMM1
#define KT1 (Dd/32)   // 16  k-tiles for GEMM1
#define NT2 (Dd/16)   // 32  n-tiles for GEMM2
#define KT2 (Ff/32)   // 64  k-tiles for GEMM2
#define FC  256       // F-chunk processed per stage
#define NCH (Ff/FC)   // 8 chunks
#define BM  32        // rows per block (2 WMMA m-stripes)

static __device__ __forceinline__ v8f wmma_bf16(v16bf a, v16bf b, v8f c) {
  // D = A(16x32 bf16) * B(32x16 bf16) + C(16x16 f32)
  return __builtin_amdgcn_wmma_f32_16x16x32_bf16(false, a, false, b,
                                                 (short)0, c, false, false);
}

// ---------------------------------------------------------------------------
// Gates + guide loss.
// ---------------------------------------------------------------------------
__global__ __launch_bounds__(64) void gates_kernel(
    const float* __restrict__ logits, const int* __restrict__ masks,
    float* __restrict__ gates, float* __restrict__ guide) {
  __shared__ float red[Bb];
  int b = threadIdx.x;
  float lg[Ee];
  float mx = -3.0e38f;
  for (int e = 0; e < Ee; ++e) { lg[e] = logits[b * Ee + e]; mx = fmaxf(mx, lg[e]); }
  float s = 0.f;
  for (int e = 0; e < Ee; ++e) { lg[e] = __expf(lg[e] - mx); s += lg[e]; }
  float gs = 0.f;
  float gt[Ee];
  for (int e = 0; e < Ee; ++e) {
    float raw = lg[e] / s;
    float gm  = (masks[b * Ee + e] == 1) ? raw : 0.f;
    gt[e] = gm; gs += gm;
  }
  float inv = 1.0f / (gs + 1e-9f);
  for (int e = 0; e < Ee; ++e) gates[b * Ee + e] = gt[e] * inv;
  red[b] = gs;
  __syncthreads();
  if (b == 0) {
    float tot = 0.f;
    for (int i = 0; i < Bb; ++i) tot += red[i];
    float gl = 1.0f - tot / (float)Bb;
    guide[0] = gl * gl;
  }
}

// ---------------------------------------------------------------------------
// Pack w[e][K][N] (row-major) into hi/lo bf16 WMMA B-operand tiles (32x16):
//   lane l holds column n = n0+(l&15), K = (l>=16 ? 8:0) + j + (j>=8 ? 8:0), j=0..15.
// Tile block = 1024 bf16: [0..511] hi (lane-major 16), [512..1023] lo.
// ---------------------------------------------------------------------------
__global__ __launch_bounds__(512) void pack_b(
    const float* __restrict__ w, __bf16* __restrict__ dst,
    int NT, int KT, int K, int N) {
  int tile = blockIdx.x;
  int t = threadIdx.x;
  int kt = tile % KT;
  int nt = (tile / KT) % NT;
  int e  = tile / (KT * NT);
  int lane = t >> 4, j = t & 15;
  int n = nt * 16 + (lane & 15);
  int k = kt * 32 + ((lane >> 4) << 3) + j + ((j >> 3) << 3);
  float v = w[((size_t)e * K + k) * (size_t)N + n];
  __bf16 hi = (__bf16)v;
  __bf16 lo = (__bf16)(v - (float)hi);
  size_t base = (size_t)tile * 1024;
  dst[base + lane * 16 + j]       = hi;
  dst[base + 512 + lane * 16 + j] = lo;
}

// ---------------------------------------------------------------------------
// Fused MoE kernel: one block (256 thr = 8 waves) per 32-row stripe of X.
// ---------------------------------------------------------------------------
__global__ __launch_bounds__(256) void moe_kernel(
    const float* __restrict__ x,
    const float* __restrict__ b1, const float* __restrict__ b2,
    const __bf16* __restrict__ w1pk, const __bf16* __restrict__ w2pk,
    const float* __restrict__ gates, float* __restrict__ out) {
  // x stripe in WMMA-A layout, hi/lo split: 2 m-stripes * 16 k-tiles * 2KB = 64KB
  __shared__ __bf16 xpk[2 * KT1 * 1024];
  // (g*gelu(h)) chunk in WMMA-A layout: 2 m-stripes * 8 k-tiles * 2KB = 32KB
  __shared__ __bf16 hpk[2 * (FC / 32) * 1024];

  const int stripe = blockIdx.x;           // 0..1023
  const int m0     = stripe * BM;
  const int batch  = stripe >> 4;          // 16 stripes per batch row
  const int t      = threadIdx.x;
  const int wave   = t >> 5;
  const int lane   = t & 31;
  const int hl     = lane >> 4;
  const int lcol   = lane & 15;

  // ---- stage x stripe into LDS in A layout (hi/lo) -------------------------
  for (int idx = t; idx < BM * Dd; idx += 256) {
    int ms = idx >> 13;                    // m-stripe 0/1
    int rem = idx & 8191;
    int kt = rem >> 9;
    int r  = rem & 511;
    int ln = r >> 4;
    int j  = r & 15;
    int m  = ln & 15;
    // A layout: lanes<16 hold K 0-7,16-23 ; lanes>=16 hold K 8-15,24-31
    int kloc = ((ln >> 4) << 3) + j + ((j >> 3) << 3);
    int k = kt * 32 + kloc;
    float v = x[(size_t)(m0 + ms * 16 + m) * Dd + k];
    __bf16 hi = (__bf16)v;
    __bf16 lo = (__bf16)(v - (float)hi);
    __bf16* dst = xpk + (ms * KT1 + kt) * 1024 + ln * 16 + j;
    dst[0]   = hi;
    dst[512] = lo;
  }
  __syncthreads();

  float garr[Ee];
  for (int e = 0; e < Ee; ++e) garr[e] = gates[batch * Ee + e];

  // grand accumulator across experts: [ms][q] -> 8 tiles of 16x16
  v8f gacc[8];
  for (int q = 0; q < 8; ++q)
    for (int i = 0; i < 8; ++i) gacc[q][i] = 0.f;

  for (int e = 0; e < Ee; ++e) {
    const float g = garr[e];
    if (g == 0.0f) continue;               // block-uniform: EXEC stays full

    for (int c = 0; c < NCH; ++c) {
      // ---- GEMM1: h_chunk[32, FC] = x(32xD) @ W1[:, c*FC .. ] -------------
      v8f hacc[4];                          // [ms*2 + nloc]
      for (int q = 0; q < 4; ++q)
        for (int i = 0; i < 8; ++i) hacc[q][i] = 0.f;
      const int nt1 = c * (FC / 16) + wave * 2;   // two 16-wide f-tiles/wave
      for (int kt = 0; kt < KT1; ++kt) {
        const v16bf* at0 = (const v16bf*)(xpk + (0 * KT1 + kt) * 1024);
        const v16bf* at1 = (const v16bf*)(xpk + (1 * KT1 + kt) * 1024);
        v16bf a0h = at0[lane], a0l = at0[32 + lane];
        v16bf a1h = at1[lane], a1l = at1[32 + lane];
        const v16bf* bt0 =
            (const v16bf*)(w1pk + (size_t)((e * NT1 + nt1) * KT1 + kt) * 1024);
        const v16bf* bt1 =
            (const v16bf*)(w1pk + (size_t)((e * NT1 + nt1 + 1) * KT1 + kt) * 1024);
        v16bf b0h = bt0[lane], b0l = bt0[32 + lane];
        v16bf b1h = bt1[lane], b1l = bt1[32 + lane];
        hacc[0] = wmma_bf16(a0h, b0h, hacc[0]);
        hacc[0] = wmma_bf16(a0h, b0l, hacc[0]);
        hacc[0] = wmma_bf16(a0l, b0h, hacc[0]);
        hacc[1] = wmma_bf16(a0h, b1h, hacc[1]);
        hacc[1] = wmma_bf16(a0h, b1l, hacc[1]);
        hacc[1] = wmma_bf16(a0l, b1h, hacc[1]);
        hacc[2] = wmma_bf16(a1h, b0h, hacc[2]);
        hacc[2] = wmma_bf16(a1h, b0l, hacc[2]);
        hacc[2] = wmma_bf16(a1l, b0h, hacc[2]);
        hacc[3] = wmma_bf16(a1h, b1h, hacc[3]);
        hacc[3] = wmma_bf16(a1h, b1l, hacc[3]);
        hacc[3] = wmma_bf16(a1l, b1h, hacc[3]);
      }

      __syncthreads();   // previous chunk's GEMM2 done reading hpk

      // ---- epilogue: +b1, exact GELU, *g, hi/lo split, scatter to A layout -
      auto stage_h = [&](const v8f& acc, int ms, int ntc) {
        int fl   = ntc * 16 + lcol;          // f within chunk
        int f    = c * FC + fl;              // global f
        float bb = b1[e * Ff + f];
        int kt2  = fl >> 5;
        int kloc = fl & 31;
        int hb   = (kloc >> 3) & 1;
        int j    = kloc - hb * 8;
        if (j >= 16) j -= 8;
        __bf16* hdst = hpk + (ms * (FC / 32) + kt2) * 1024 + j;
        for (int v = 0; v < 8; ++v) {
          float hv = acc[v] + bb;
          float ge = 0.5f * hv * (1.0f + erff(hv * 0.70710678118654752f));
          float sg = g * ge;                 // fold gate into h
          __bf16 hi = (__bf16)sg;
          __bf16 lo = (__bf16)(sg - (float)hi);
          int ld = (v + 8 * hl) + 16 * hb;
          hdst[ld * 16]       = hi;
          hdst[512 + ld * 16] = lo;
        }
      };
      stage_h(hacc[0], 0, wave * 2);
      stage_h(hacc[1], 0, wave * 2 + 1);
      stage_h(hacc[2], 1, wave * 2);
      stage_h(hacc[3], 1, wave * 2 + 1);
      __syncthreads();

      // ---- GEMM2 partial: gacc += (g*h)(32xFC) @ W2[chunk, :] -------------
      for (int ktl = 0; ktl < FC / 32; ++ktl) {
        const v16bf* at0 = (const v16bf*)(hpk + (0 * (FC / 32) + ktl) * 1024);
        const v16bf* at1 = (const v16bf*)(hpk + (1 * (FC / 32) + ktl) * 1024);
        v16bf a0h = at0[lane], a0l = at0[32 + lane];
        v16bf a1h = at1[lane], a1l = at1[32 + lane];
        int kt2g = c * (FC / 32) + ktl;
#pragma unroll
        for (int q = 0; q < 4; ++q) {
          int nt2 = wave * 4 + q;            // four 16-wide d-tiles/wave
          const v16bf* bt =
              (const v16bf*)(w2pk + (size_t)((e * NT2 + nt2) * KT2 + kt2g) * 1024);
          v16bf bh = bt[lane], bl = bt[32 + lane];
          gacc[q] = wmma_bf16(a0h, bh, gacc[q]);
          gacc[q] = wmma_bf16(a0h, bl, gacc[q]);
          gacc[q] = wmma_bf16(a0l, bh, gacc[q]);
          gacc[4 + q] = wmma_bf16(a1h, bh, gacc[4 + q]);
          gacc[4 + q] = wmma_bf16(a1h, bl, gacc[4 + q]);
          gacc[4 + q] = wmma_bf16(a1l, bh, gacc[4 + q]);
        }
      }
    }  // chunks
  }  // experts

  // ---- store output stripe: out = gacc + sum_e g_e*b2[e] -------------------
#pragma unroll
  for (int q = 0; q < 4; ++q) {
    int d = (wave * 4 + q) * 16 + lcol;
    float bs = 0.f;
    for (int e = 0; e < Ee; ++e) bs += garr[e] * b2[e * Dd + d];
    for (int ms = 0; ms < 2; ++ms) {
      for (int v = 0; v < 8; ++v) {
        int m = ms * 16 + v + 8 * hl;
        out[(size_t)(m0 + m) * Dd + d] = gacc[ms * 4 + q][v] + bs;
      }
    }
  }
}

// ---------------------------------------------------------------------------
extern "C" void kernel_launch(void* const* d_in, const int* in_sizes, int n_in,
                              void* d_out, int out_size, void* d_ws, size_t ws_size,
                              hipStream_t stream) {
  const float* x      = (const float*)d_in[0];  // [B,L,D]
  const float* logits = (const float*)d_in[1];  // [B,E]
  const int*   masks  = (const int*)d_in[2];    // [B,E]
  const float* w1     = (const float*)d_in[3];  // [E,D,F]
  const float* b1     = (const float*)d_in[4];  // [E,F]
  const float* w2     = (const float*)d_in[5];  // [E,F,D]
  const float* b2     = (const float*)d_in[6];  // [E,D]

  float* out   = (float*)d_out;
  float* guide = out + (size_t)Bb * Ll * Dd;

  // Workspace: gates (4KB pad) | w1pk (33.5MB) | w2pk (33.5MB)
  const size_t W1PK_ELEMS = (size_t)Ee * NT1 * KT1 * 1024;
  const size_t W2PK_ELEMS = (size_t)Ee * NT2 * KT2 * 1024;
  char* wsb = (char*)d_ws;
  float*  gates = (float*)wsb;
  __bf16* w1pk  = (__bf16*)(wsb + 4096);
  __bf16* w2pk  = (__bf16*)(wsb + 4096 + W1PK_ELEMS * sizeof(__bf16));
  size_t need = 4096 + (W1PK_ELEMS + W2PK_ELEMS) * sizeof(__bf16);
  if (ws_size < need) return;  // ~67MB scratch required

  gates_kernel<<<1, Bb, 0, stream>>>(logits, masks, gates, guide);
  pack_b<<<Ee * NT1 * KT1, 512, 0, stream>>>(w1, w1pk, NT1, KT1, Dd, Ff);
  pack_b<<<Ee * NT2 * KT2, 512, 0, stream>>>(w2, w2pk, NT2, KT2, Ff, Dd);

  const int stripes = (Bb * Ll) / BM;  // 1024
  moe_kernel<<<stripes, 256, 0, stream>>>(x, b1, b2, w1pk, w2pk, gates, out);
}
